// WaveNet_90159953478035
// MI455X (gfx1250) — matrix operand order repaired and checked by reading, not compiled, not run on hardware
//
#include <hip/hip_runtime.h>

#define BB 4
#define TT 32768
#define NMEL 80
#define RC 120
#define SCC 240
#define NBLK 16
#define NCLS 256

#define CP_RES 128
#define CP_COND 96
#define CP_SKIP 256

// per-block packed weight slab (u16 elements)
#define OFF_W0 0
#define OFF_W1 32768
#define OFF_WC 65536
#define OFF_WS 90112
#define OFF_WR 122880
#define WBLK_ELEMS 139264   // = 272KB

typedef unsigned short u16;
typedef __attribute__((ext_vector_type(16))) __bf16 v16bf;
typedef __attribute__((ext_vector_type(8)))  float  v8f;

union FragB { v16bf v; u16 h[16]; uint4 q[2]; };
union FragC { v8f   v; float f[8]; uint4 q[2]; };

__device__ __forceinline__ u16 f2bf(float f) {
    unsigned u = __float_as_uint(f);
    unsigned r = u + 0x7fffu + ((u >> 16) & 1u);
    return (u16)(r >> 16);
}
__device__ __forceinline__ float bf2f(u16 h) { return __uint_as_float(((unsigned)h) << 16); }

__device__ __forceinline__ v16bf ld16(const u16* p) {
    FragB f; f.q[0] = *(const uint4*)p; f.q[1] = *(const uint4*)(p + 8); return f.v;
}
__device__ __forceinline__ v8f ld8f(const float* p) {
    FragC c; c.q[0] = *(const uint4*)p; c.q[1] = *(const uint4*)(p + 4); return c.v;
}
__device__ __forceinline__ void st8f(float* p, v8f x) {
    FragC c; c.v = x; *(uint4*)p = c.q[0]; *(uint4*)(p + 4) = c.q[1];
}
__device__ __forceinline__ void st8bf(u16* p, v8f x) {
    FragC c; c.v = x; FragB b;
#pragma unroll
    for (int i = 0; i < 8; i++) b.h[i] = f2bf(c.f[i]);
    *(uint4*)p = b.q[0];
}

__device__ __forceinline__ v8f wmma_bf(v16bf a, v16bf b, v8f c) {
    return __builtin_amdgcn_wmma_f32_16x16x32_bf16(false, a, false, b, (short)0, c, false, false);
}

// B fragment: lane n=lane&15 -> column, 16 consecutive K starting at kbase+((lane>>4)*16)
__device__ __forceinline__ v16bf ldB(const u16* colbase, int cp, int kbase, int lane) {
    int n = lane & 15, ks = (lane >> 4) << 4;
    return ld16(colbase + n * cp + kbase + ks);
}
__device__ __forceinline__ v16bf ldBshift(const u16* batchbase, int cp, int tbase, int d,
                                          int kbase, int lane) {
    int n = lane & 15, ks = (lane >> 4) << 4;
    int t = tbase + n - d;
    if (t < 0) { FragB z; z.q[0] = make_uint4(0,0,0,0); z.q[1] = z.q[0]; return z.v; }
    return ld16(batchbase + (size_t)t * cp + kbase + ks);
}
// A fragment from packed weights: 512 halves per fragment, lane-major
__device__ __forceinline__ v16bf ldA(const u16* W, int nK, int rt, int kc, int lane) {
    return ld16(W + (((size_t)(rt * nK + kc)) << 9) + lane * 16);
}
// C/D fragments in [t][c] layout
__device__ __forceinline__ v8f ldCf(const float* colbase, int cp, int rt, int lane) {
    int n = lane & 15, mb = (lane >> 4) << 3;
    return ld8f(colbase + n * cp + rt * 16 + mb);
}
__device__ __forceinline__ v8f ldCbias(const float* bias, int rt, int lane) {
    int mb = (lane >> 4) << 3;
    return ld8f(bias + rt * 16 + mb);
}
__device__ __forceinline__ v8f ldCbf(const u16* colbase, int cp, int rt, int lane) {
    int n = lane & 15, mb = (lane >> 4) << 3;
    const u16* p = colbase + n * cp + rt * 16 + mb;
    FragB b; b.q[0] = *(const uint4*)p;
    FragC c;
#pragma unroll
    for (int i = 0; i < 8; i++) c.f[i] = bf2f(b.h[i]);
    return c.v;
}
__device__ __forceinline__ void stCf(float* colbase, int cp, int rt, int lane, v8f x) {
    int n = lane & 15, mb = (lane >> 4) << 3;
    st8f(colbase + n * cp + rt * 16 + mb, x);
}
__device__ __forceinline__ void stCbf(u16* colbase, int cp, int rt, int lane, v8f x) {
    int n = lane & 15, mb = (lane >> 4) << 3;
    st8bf(colbase + n * cp + rt * 16 + mb, x);
}

__device__ __forceinline__ float sigf(float x) { return 1.f / (1.f + __expf(-x)); }

// ---------------- prologue kernels ----------------

__global__ void init_res(const float* __restrict__ wav, const float* __restrict__ wav_w,
                         const float* __restrict__ wav_b, u16* __restrict__ resH) {
    int idx = blockIdx.x * blockDim.x + threadIdx.x;
    if (idx >= BB * TT * CP_RES) return;
    int c = idx & 127, bt = idx >> 7;
    float v = 0.f;
    if (c < RC) v = wav_w[c] * wav[bt] + wav_b[c];
    resH[idx] = f2bf(v);
}

__global__ void init_cond(const float* __restrict__ cond, u16* __restrict__ condH) {
    int idx = blockIdx.x * blockDim.x + threadIdx.x;
    if (idx >= BB * TT * CP_COND) return;
    int k = idx % CP_COND, bt = idx / CP_COND;
    int b = bt >> 15, t = bt & (TT - 1);
    float v = (k < NMEL) ? cond[((size_t)b * NMEL + k) * TT + t] : 0.f;
    condH[idx] = f2bf(v);
}

// pack weights (O x K) into WMMA A-fragment order; optional gating row split
__global__ void pack_w(const float* __restrict__ src, u16* __restrict__ dst,
                       int base, int rowStride, int kStride,
                       int srcO, int srcK, int dstO, int dstK, int gate) {
    int e = blockIdx.x * blockDim.x + threadIdx.x;
    if (e >= dstO * dstK) return;
    int h = e & 15, lane = (e >> 4) & 31, frag = e >> 9;
    int nK = dstK >> 5;
    int rt = frag / nK, kc = frag % nK;
    int m = lane & 15;
    int k = ((lane >> 4) << 4) | h;
    int M = rt * 16 + m, K = kc * 32 + k;
    int row;
    if (gate) row = (M < 120) ? M : (M >= 128 && M < 248) ? (M - 8) : -1;
    else      row = (M < srcO) ? M : -1;
    float v = (row >= 0 && K < srcK) ? src[base + row * rowStride + K * kStride] : 0.f;
    dst[e] = f2bf(v);
}

__global__ void pack_bias(const float* __restrict__ dil_b, const float* __restrict__ cond_b,
                          const float* __restrict__ skip_b, const float* __restrict__ res_b,
                          float* __restrict__ gbias, float* __restrict__ skipb,
                          float* __restrict__ resb) {
    int e = blockIdx.x * blockDim.x + threadIdx.x;
    if (e >= NBLK * 640) return;
    int i = e / 640, s = e % 640;
    if (s < 256) {
        int r = s;
        int row = (r < 120) ? r : (r >= 128 && r < 248) ? (r - 8) : -1;
        gbias[i * 256 + r] = (row >= 0) ? dil_b[i * 240 + row] + cond_b[i * 240 + row] : 0.f;
    } else if (s < 512) {
        int r = s - 256;
        skipb[i * 256 + r] = (r < 240) ? skip_b[i * 240 + r] : 0.f;
    } else {
        int r = s - 512;
        resb[i * 128 + r] = (r < 120) ? res_b[i * 120 + r] : 0.f;
    }
}

// ---------------- fused residual block ----------------
// 8 waves / workgroup; block weights staged in LDS (shared by all waves);
// each wave processes 2 sixteen-column time tiles.
__global__ __launch_bounds__(256) void wnet_block(
    const u16* __restrict__ condH, const u16* __restrict__ resIn,
    u16* __restrict__ resOut, float* __restrict__ skipF,
    const u16* __restrict__ Wblk,
    const float* __restrict__ gbias, const float* __restrict__ skipb,
    const float* __restrict__ resb, int d, int first) {
    __shared__ u16 wlds[WBLK_ELEMS];         // 272KB packed weights
    __shared__ u16 stage[8 * 16 * CP_RES];   // 32KB D->B transpose staging
    int lane = threadIdx.x & 31, wid = threadIdx.x >> 5;

    // cooperative weight staging: 272KB global -> LDS, once per workgroup
    {
        const uint4* src = (const uint4*)Wblk;
        uint4* dst = (uint4*)wlds;
        for (int q = threadIdx.x; q < WBLK_ELEMS / 8; q += 256) dst[q] = src[q];
    }
    __syncthreads();
    const u16* W0 = wlds + OFF_W0;
    const u16* W1 = wlds + OFF_W1;
    const u16* Wc = wlds + OFF_WC;
    const u16* Ws = wlds + OFF_WS;
    const u16* Wr = wlds + OFF_WR;
    u16* my = stage + wid * (16 * CP_RES);

    for (int tt = 0; tt < 2; tt++) {
        int tile = (blockIdx.x * 8 + wid) * 2 + tt;   // 0..8191
        int b = tile >> 11;                           // 2048 tiles per batch
        int tbase = (tile & 2047) << 4;
        const u16* resBatch = resIn + (size_t)b * TT * CP_RES;
        const u16* resCol   = resBatch + (size_t)tbase * CP_RES;
        const u16* condCol  = condH + ((size_t)b * TT + tbase) * CP_COND;
        float*     skipCol  = skipF + ((size_t)b * TT + tbase) * CP_SKIP;
        u16*       resOCol  = resOut + ((size_t)b * TT + tbase) * CP_RES;

        v16bf bx[4], bs[4], bc[3];
#pragma unroll
        for (int kc = 0; kc < 4; kc++) {
            bx[kc] = ldB(resCol, CP_RES, kc * 32, lane);
            bs[kc] = ldBshift(resBatch, CP_RES, tbase, d, kc * 32, lane);
        }
#pragma unroll
        for (int kc = 0; kc < 3; kc++) bc[kc] = ldB(condCol, CP_COND, kc * 32, lane);

        // g = W0@x[t-d] + W1@x[t] + Wc@cond + bias; rows [0,128)=tanh, [128,256)=sigmoid
#pragma unroll
        for (int j = 0; j < 8; j++) {
            v8f a0 = ldCbias(gbias, j, lane);
            v8f a1 = ldCbias(gbias, j + 8, lane);
#pragma unroll
            for (int kc = 0; kc < 4; kc++) {
                a0 = wmma_bf(ldA(W0, 4, j, kc, lane), bs[kc], a0);
                a1 = wmma_bf(ldA(W0, 4, j + 8, kc, lane), bs[kc], a1);
                a0 = wmma_bf(ldA(W1, 4, j, kc, lane), bx[kc], a0);
                a1 = wmma_bf(ldA(W1, 4, j + 8, kc, lane), bx[kc], a1);
            }
#pragma unroll
            for (int kc = 0; kc < 3; kc++) {
                a0 = wmma_bf(ldA(Wc, 3, j, kc, lane), bc[kc], a0);
                a1 = wmma_bf(ldA(Wc, 3, j + 8, kc, lane), bc[kc], a1);
            }
            FragC ca, cb, co; ca.v = a0; cb.v = a1;
#pragma unroll
            for (int r = 0; r < 8; r++) {
                float th = 2.f * sigf(2.f * ca.f[r]) - 1.f;   // tanh
                co.f[r] = th * sigf(cb.f[r]);
            }
            stCbf(my, CP_RES, j, lane, co.v);                 // stage D->B transpose
        }
        asm volatile("s_wait_dscnt 0" ::: "memory");

        v16bf bo[4];
#pragma unroll
        for (int kc = 0; kc < 4; kc++) bo[kc] = ldB(my, CP_RES, kc * 32, lane);

        // skip_sum += Ws @ out + skip_b
#pragma unroll
        for (int rt = 0; rt < 16; rt++) {
            v8f acc = ldCbias(skipb, rt, lane);
            if (!first) acc = acc + ldCf(skipCol, CP_SKIP, rt, lane);
#pragma unroll
            for (int kc = 0; kc < 4; kc++) acc = wmma_bf(ldA(Ws, 4, rt, kc, lane), bo[kc], acc);
            stCf(skipCol, CP_SKIP, rt, lane, acc);
        }
        // res = Wr @ out + res_b + res_old
#pragma unroll
        for (int rt = 0; rt < 8; rt++) {
            v8f acc = ldCbias(resb, rt, lane);
            acc = acc + ldCbf(resCol, CP_RES, rt, lane);
#pragma unroll
            for (int kc = 0; kc < 4; kc++) acc = wmma_bf(ldA(Wr, 4, rt, kc, lane), bo[kc], acc);
            stCbf(resOCol, CP_RES, rt, lane, acc);
        }
    }
}

// ---------------- fused output head ----------------
__global__ __launch_bounds__(128) void wnet_final(
    const float* __restrict__ skipF, const u16* __restrict__ Wo, const u16* __restrict__ We,
    const float* __restrict__ out_b, const float* __restrict__ end_b,
    float* __restrict__ out) {
    __shared__ u16 lds[4 * 16 * 256];
    int lane = threadIdx.x & 31, wid = threadIdx.x >> 5;
    int gw = blockIdx.x * 4 + wid;
    int b = gw >> 11;
    int tbase = (gw & 2047) << 4;
    const float* skipCol = skipF + ((size_t)b * TT + tbase) * CP_SKIP;
    u16* my = lds + wid * (16 * 256);
    int n = lane & 15, ks = (lane >> 4) << 4, mb = (lane >> 4) << 3;

    // relu(skip) -> bf16 B fragments (lane-local, no shuffle needed)
    v16bf bsk[8];
#pragma unroll
    for (int kc = 0; kc < 8; kc++) {
        const float* p = skipCol + n * CP_SKIP + kc * 32 + ks;
        FragB f;
#pragma unroll
        for (int i = 0; i < 16; i++) { float x = p[i]; f.h[i] = f2bf(x > 0.f ? x : 0.f); }
        bsk[kc] = f.v;
    }
    // o = relu(Wo @ relu(skip) + out_b), staged to LDS for D->B transpose
#pragma unroll
    for (int rt = 0; rt < 16; rt++) {
        v8f acc = ldCbias(out_b, rt, lane);
#pragma unroll
        for (int kc = 0; kc < 8; kc++) acc = wmma_bf(ldA(Wo, 8, rt, kc, lane), bsk[kc], acc);
        FragC c; c.v = acc; FragB h;
#pragma unroll
        for (int r = 0; r < 8; r++) { float x = c.f[r]; h.h[r] = f2bf(x > 0.f ? x : 0.f); }
        *(uint4*)(my + n * 256 + rt * 16 + mb) = h.q[0];
    }
    asm volatile("s_wait_dscnt 0" ::: "memory");

    v16bf bo[8];
#pragma unroll
    for (int kc = 0; kc < 8; kc++) bo[kc] = ldB(my, 256, kc * 32, lane);
    // end = We @ o + end_b, scatter to [b][c][t]
#pragma unroll
    for (int rt = 0; rt < 16; rt++) {
        v8f acc = ldCbias(end_b, rt, lane);
#pragma unroll
        for (int kc = 0; kc < 8; kc++) acc = wmma_bf(ldA(We, 8, rt, kc, lane), bo[kc], acc);
        FragC c; c.v = acc;
        float* op = out + ((size_t)b * NCLS + rt * 16 + mb) * TT + (tbase + n);
#pragma unroll
        for (int r = 0; r < 8; r++) op[(size_t)r * TT] = c.f[r];
    }
}

extern "C" void kernel_launch(void* const* d_in, const int* in_sizes, int n_in,
                              void* d_out, int out_size, void* d_ws, size_t ws_size,
                              hipStream_t stream) {
    (void)in_sizes; (void)n_in; (void)out_size; (void)ws_size;
    const float* wav    = (const float*)d_in[0];
    const float* cond   = (const float*)d_in[1];
    const float* wav_w  = (const float*)d_in[2];
    const float* wav_b  = (const float*)d_in[3];
    const float* cond_w = (const float*)d_in[4];
    const float* cond_b = (const float*)d_in[5];
    const float* dil_w  = (const float*)d_in[6];
    const float* dil_b  = (const float*)d_in[7];
    const float* skip_w = (const float*)d_in[8];
    const float* skip_b = (const float*)d_in[9];
    const float* res_w  = (const float*)d_in[10];
    const float* res_b  = (const float*)d_in[11];
    const float* out_w  = (const float*)d_in[12];
    const float* out_b  = (const float*)d_in[13];
    const float* end_w  = (const float*)d_in[14];
    const float* end_b  = (const float*)d_in[15];

    char* ws = (char*)d_ws;
    size_t off = 0;
    auto alloc = [&](size_t bytes) {
        size_t p = off; off += (bytes + 255) & ~(size_t)255; return p;
    };
    u16*   resA  = (u16*)(ws + alloc((size_t)BB * TT * CP_RES * 2));
    u16*   resBf = (u16*)(ws + alloc((size_t)BB * TT * CP_RES * 2));
    float* skipF = (float*)(ws + alloc((size_t)BB * TT * CP_SKIP * 4));
    u16*   condH = (u16*)(ws + alloc((size_t)BB * TT * CP_COND * 2));
    u16*   blkW  = (u16*)(ws + alloc((size_t)NBLK * WBLK_ELEMS * 2));
    u16*   WoP   = (u16*)(ws + alloc((size_t)256 * 256 * 2));
    u16*   WeP   = (u16*)(ws + alloc((size_t)256 * 256 * 2));
    float* gbias  = (float*)(ws + alloc((size_t)NBLK * 256 * 4));
    float* skipbP = (float*)(ws + alloc((size_t)NBLK * 256 * 4));
    float* resbP  = (float*)(ws + alloc((size_t)NBLK * 128 * 4));

    auto cdiv = [](int a, int b) { return (a + b - 1) / b; };

    init_res<<<cdiv(BB * TT * CP_RES, 256), 256, 0, stream>>>(wav, wav_w, wav_b, resA);
    init_cond<<<cdiv(BB * TT * CP_COND, 256), 256, 0, stream>>>(cond, condH);
    pack_bias<<<cdiv(NBLK * 640, 256), 256, 0, stream>>>(dil_b, cond_b, skip_b, res_b,
                                                         gbias, skipbP, resbP);
    for (int i = 0; i < NBLK; i++) {
        u16* base = blkW + (size_t)i * WBLK_ELEMS;
        // dil_w: [NB][240][120][2]
        pack_w<<<cdiv(256 * 128, 256), 256, 0, stream>>>(dil_w, base + OFF_W0,
            i * 240 * 120 * 2, 240, 2, 240, 120, 256, 128, 1);
        pack_w<<<cdiv(256 * 128, 256), 256, 0, stream>>>(dil_w, base + OFF_W1,
            i * 240 * 120 * 2 + 1, 240, 2, 240, 120, 256, 128, 1);
        pack_w<<<cdiv(256 * 96, 256), 256, 0, stream>>>(cond_w, base + OFF_WC,
            i * 240 * 80, 80, 1, 240, 80, 256, 96, 1);
        pack_w<<<cdiv(256 * 128, 256), 256, 0, stream>>>(skip_w, base + OFF_WS,
            i * 240 * 120, 120, 1, 240, 120, 256, 128, 0);
        pack_w<<<cdiv(128 * 128, 256), 256, 0, stream>>>(res_w, base + OFF_WR,
            i * 120 * 120, 120, 1, 120, 120, 128, 128, 0);
    }
    pack_w<<<cdiv(256 * 256, 256), 256, 0, stream>>>(out_w, WoP, 0, 240, 1, 256, 240, 256, 256, 0);
    pack_w<<<cdiv(256 * 256, 256), 256, 0, stream>>>(end_w, WeP, 0, 256, 1, 256, 256, 256, 256, 0);

    u16* rin = resA; u16* rout = resBf;
    for (int i = 0; i < NBLK; i++) {
        int d = 1 << (i & 7);
        wnet_block<<<512, 256, 0, stream>>>(condH, rin, rout, skipF,
            blkW + (size_t)i * WBLK_ELEMS,
            gbias + i * 256, skipbP + i * 256, resbP + i * 128, d, i == 0 ? 1 : 0);
        u16* t = rin; rin = rout; rout = t;
    }
    wnet_final<<<2048, 128, 0, stream>>>(skipF, WoP, WeP, out_b, end_b, (float*)d_out);
}
